// SelfAttention_26671746908779
// MI455X (gfx1250) — matrix operand order repaired and checked by reading, not compiled, not run on hardware
//
#include <hip/hip_runtime.h>
#include <hip/hip_bf16.h>

typedef __attribute__((ext_vector_type(16))) __bf16 v16bf;
typedef __attribute__((ext_vector_type(8)))  float  v8f;

#define NPIX 4096   // H*W
#define CCH  256    // channels
#define CK   32     // C/8
#define XSTR 264    // LDS x-tile row stride in bf16 elems (528B, 132 dwords)

union BF16x16 { v16bf v; uint4 q[2]; __bf16 e[16]; };
union BF16x8  { uint4 q;  __bf16 e[8]; };

static __device__ __forceinline__ v8f wmma_bf16f32(v16bf a, v16bf b, v8f c) {
  // D = A(16x32 bf16) x B(32x16 bf16) + C(16x16 f32)
  return __builtin_amdgcn_wmma_f32_16x16x32_bf16(false, a, false, b, (short)0, c, false, false);
}

// ---------------------------------------------------------------------------
// Stage 0: pack [wq; wk; wv] into one bf16 weight matrix Wb[320][256] so the
// projection GEMM reads A-tiles as b128 loads with no conversion.
// ---------------------------------------------------------------------------
__global__ __launch_bounds__(256) void wcvt_kernel(
    const float* __restrict__ wq, const float* __restrict__ wk,
    const float* __restrict__ wv, __bf16* __restrict__ Wb)
{
  const int i = blockIdx.x * 256 + threadIdx.x;    // 320*256 elements
  const int o = i >> 8, c = i & 255;
  const float* w = (o < CK)     ? (wq + o * CCH)
                 : (o < 2 * CK) ? (wk + (o - CK) * CCH)
                                : (wv + (o - 2 * CK) * CCH);
  Wb[i] = (__bf16)w[c];
}

// ---------------------------------------------------------------------------
// Stage 1: WMMA projection GEMM. D[320][4096] = Wb x X per batch.
// Block = (batch, 64-pixel tile), 10 waves; wave w owns rows [32w, 32w+32)
// (2 A-tiles) x 4 column subtiles. X tile staged bf16-transposed in LDS so
// B-tiles are two ds_load_b128 in native WMMA B layout.
// Rows 0..31 -> fT[b][n][32], 32..63 -> gT[b][m][32], 64..319 -> h[b][c][n].
// ---------------------------------------------------------------------------
__global__ __launch_bounds__(320) void proj_kernel(
    const float* __restrict__ x, const __bf16* __restrict__ Wb,
    __bf16* __restrict__ fT, __bf16* __restrict__ gT, __bf16* __restrict__ h)
{
  __shared__ __align__(16) __bf16 xsb[64 * XSTR];  // ~33KB
  const int b  = blockIdx.x >> 6;
  const int n0 = (blockIdx.x & 63) * 64;
  const float* xb = x + (size_t)b * CCH * NPIX;

  for (int idx = threadIdx.x; idx < CCH * 64; idx += 320) {
    const int c = idx >> 6, n = idx & 63;            // coalesced global read
    xsb[n * XSTR + c] = (__bf16)xb[(size_t)c * NPIX + n0 + n];
  }
  __syncthreads();

  const int wave = threadIdx.x >> 5;                 // 0..9
  const int l    = threadIdx.x & 31;
  const int li   = l & 15;
  const int hi   = l >> 4;
  const int o0   = wave * 32;

  v8f acc[2][4];
#pragma unroll
  for (int t = 0; t < 2; ++t)
#pragma unroll
    for (int cs = 0; cs < 4; ++cs) acc[t][cs] = (v8f){};

  for (int kc = 0; kc < CCH; kc += 32) {
    BF16x16 A[2];
#pragma unroll
    for (int t = 0; t < 2; ++t) {
      const __bf16* wr = Wb + (size_t)(o0 + t * 16 + li) * CCH + kc;
      A[t].q[0] = *(const uint4*)(wr + hi * 8);
      A[t].q[1] = *(const uint4*)(wr + 16 + hi * 8);
    }
#pragma unroll
    for (int cs = 0; cs < 4; ++cs) {
      BF16x16 Bt;   // 32x16: lane = col n, q[0] holds K(c) kc+8*hi+0..7, q[1] kc+16+8*hi+0..7
      const uint4* xp = (const uint4*)(&xsb[(cs * 16 + li) * XSTR + kc + hi * 8]);
      Bt.q[0] = xp[0]; Bt.q[1] = xp[2];
      acc[0][cs] = wmma_bf16f32(A[0].v, Bt.v, acc[0][cs]);
      acc[1][cs] = wmma_bf16f32(A[1].v, Bt.v, acc[1][cs]);
    }
  }

#pragma unroll
  for (int t = 0; t < 2; ++t) {
#pragma unroll
    for (int cs = 0; cs < 4; ++cs) {
#pragma unroll
      for (int r = 0; r < 8; ++r) {
        const __bf16 v = (__bf16)acc[t][cs][r];
        const int o = o0 + t * 16 + r + 8 * hi;
        const int n = n0 + cs * 16 + li;
        if (o < CK)          fT[((size_t)b * NPIX + n) * CK + o] = v;
        else if (o < 2 * CK) gT[((size_t)b * NPIX + n) * CK + (o - CK)] = v;
        else                 h [((size_t)b * CCH + (o - 2 * CK)) * NPIX + n] = v;
      }
    }
  }
}

// ---------------------------------------------------------------------------
// Stage 2: flash-attention over keys n. Block = (batch, 16 output columns m).
// 9 waves: wave 8 produces S = fT x g (2 WMMA / step) + online softmax into
// double-buffered LDS; waves 0..7 own 32 value-rows each and accumulate
// o += h x P with 2 WMMA / step. Epilogue: out = gamma * o / L + x.
// ---------------------------------------------------------------------------
__global__ __launch_bounds__(288) void attn_kernel(
    const __bf16* __restrict__ fT, const __bf16* __restrict__ gT,
    const __bf16* __restrict__ hW, const float* __restrict__ x,
    const float* __restrict__ gamma, float* __restrict__ out)
{
  __shared__ __align__(16) unsigned char Pbuf[2][16 * 80];
  __shared__ float scale_lds[2][16];
  __shared__ float lsum_lds[16];

  const int b    = blockIdx.x >> 8;
  const int m0   = (blockIdx.x & 255) * 16;
  const int tid  = threadIdx.x;
  const int wave = tid >> 5;
  const int l    = tid & 31;
  const int li   = l & 15;
  const int hi   = l >> 4;

  v8f acc0 = {}; v8f acc1 = {};
  v16bf gB;
  float Mreg = -INFINITY, Lreg = 0.f;
  const __bf16* fTb = fT + (size_t)b * NPIX * CK;
  const __bf16* hWb = hW + (size_t)b * CCH * NPIX;
  const int c0 = wave * 32;

  if (wave == 8) {
    BF16x16 g;
    const __bf16* gp = gT + ((size_t)b * NPIX + m0 + li) * CK + hi * 8;
    g.q[0] = *(const uint4*)(gp);
    g.q[1] = *(const uint4*)(gp + 16);
    gB = g.v;
  }

  for (int it = 0; it < NPIX / 32; ++it) {
    const int n0 = it * 32;
    const int bi = it & 1;
    BF16x16 hA0, hA1;
    if (wave == 8) {
      BF16x16 a0, a1;
      const __bf16* f0 = fTb + (size_t)(n0 + li) * CK;
      const __bf16* f1 = fTb + (size_t)(n0 + 16 + li) * CK;
      a0.q[0] = *(const uint4*)(f0 + hi * 8);
      a0.q[1] = *(const uint4*)(f0 + 16 + hi * 8);
      a1.q[0] = *(const uint4*)(f1 + hi * 8);
      a1.q[1] = *(const uint4*)(f1 + 16 + hi * 8);
      v8f zero = {};
      v8f s0 = wmma_bf16f32(a0.v, gB, zero);
      v8f s1 = wmma_bf16f32(a1.v, gB, zero);
      float lm = s0[0];
#pragma unroll
      for (int r = 1; r < 8; ++r) lm = fmaxf(lm, s0[r]);
#pragma unroll
      for (int r = 0; r < 8; ++r) lm = fmaxf(lm, s1[r]);
      lm = fmaxf(lm, __shfl_xor(lm, 16, 32));
      const float newM = fmaxf(Mreg, lm);
      const float sc   = __expf(Mreg - newM);
      Mreg = newM;
      BF16x8 p0, p1; float ps = 0.f;
#pragma unroll
      for (int r = 0; r < 8; ++r) { const float e = __expf(s0[r] - newM); ps += e; p0.e[r] = (__bf16)e; }
#pragma unroll
      for (int r = 0; r < 8; ++r) { const float e = __expf(s1[r] - newM); ps += e; p1.e[r] = (__bf16)e; }
      ps += __shfl_xor(ps, 16, 32);
      Lreg = Lreg * sc + ps;
      *(uint4*)(&Pbuf[bi][li * 80 + hi * 16])      = p0.q;
      *(uint4*)(&Pbuf[bi][li * 80 + 32 + hi * 16]) = p1.q;
      if (l < 16) scale_lds[bi][li] = sc;
    } else {
      const __bf16* h0 = hWb + (size_t)(c0 + li) * NPIX + n0;
      const __bf16* h1 = hWb + (size_t)(c0 + 16 + li) * NPIX + n0;
      hA0.q[0] = *(const uint4*)(h0 + hi * 8);
      hA0.q[1] = *(const uint4*)(h0 + 16 + hi * 8);
      hA1.q[0] = *(const uint4*)(h1 + hi * 8);
      hA1.q[1] = *(const uint4*)(h1 + 16 + hi * 8);
      if (it + 1 < NPIX / 32) __builtin_prefetch(h0 + 32, 0, 1);
    }
    __syncthreads();
    if (wave < 8) {
      const float sc = scale_lds[bi][li];
      BF16x16 Pb;
      const uint4* pp = (const uint4*)(&Pbuf[bi][li * 80 + hi * 16]);
      Pb.q[0] = pp[0]; Pb.q[1] = pp[2];
#pragma unroll
      for (int r = 0; r < 8; ++r) { acc0[r] *= sc; acc1[r] *= sc; }
      acc0 = wmma_bf16f32(hA0.v, Pb.v, acc0);
      acc1 = wmma_bf16f32(hA1.v, Pb.v, acc1);
    }
  }

  if (wave == 8 && l < 16) lsum_lds[li] = Lreg;
  __syncthreads();

  if (wave < 8) {
    const float invL = 1.0f / lsum_lds[li];
    const float gam  = gamma[0];
    const float* xb  = x   + (size_t)b * CCH * NPIX;
    float*       ob  = out + (size_t)b * CCH * NPIX;
#pragma unroll
    for (int r = 0; r < 8; ++r) {
      const int c   = c0 + hi * 8 + r;
      size_t    idx = (size_t)c * NPIX + m0 + li;
      ob[idx] = gam * acc0[r] * invL + xb[idx];
      idx += (size_t)16 * NPIX;
      ob[idx] = gam * acc1[r] * invL + xb[idx];
    }
  }
}

extern "C" void kernel_launch(void* const* d_in, const int* in_sizes, int n_in,
                              void* d_out, int out_size, void* d_ws, size_t ws_size,
                              hipStream_t stream) {
  (void)in_sizes; (void)n_in; (void)out_size; (void)ws_size;
  const float* x     = (const float*)d_in[0];
  const float* wq    = (const float*)d_in[1];
  const float* wk    = (const float*)d_in[2];
  const float* wv    = (const float*)d_in[3];
  const float* gamma = (const float*)d_in[4];
  float* out = (float*)d_out;

  // workspace: fT (1MB) | gT (1MB) | h (8MB) | Wb (160KB), all bf16
  __bf16* fT = (__bf16*)d_ws;
  __bf16* gT = fT + (size_t)4 * NPIX * CK;
  __bf16* hW = gT + (size_t)4 * NPIX * CK;
  __bf16* Wb = hW + (size_t)4 * CCH * NPIX;

  wcvt_kernel<<<dim3(320), dim3(256), 0, stream>>>(wq, wk, wv, Wb);
  proj_kernel<<<dim3(4 * 64), dim3(320), 0, stream>>>(x, Wb, fT, gT, hW);
  attn_kernel<<<dim3(4 * 256), dim3(288), 0, stream>>>(fT, gT, hW, x, gamma, out);
}
